// GatedLinearAttention_58377195487403
// MI455X (gfx1250) — matrix operand order repaired and checked by reading, not compiled, not run on hardware
//
#include <hip/hip_runtime.h>
#include <cstdint>
#include <cstddef>

// ---------------------------------------------------------------------------
// CDNA5 (gfx1250, wave32) WMMA types
// ---------------------------------------------------------------------------
typedef _Float16 h16 __attribute__((ext_vector_type(16)));
typedef _Float16 h8v __attribute__((ext_vector_type(8)));
typedef float    f8v __attribute__((ext_vector_type(8)));

namespace {
constexpr int kB  = 2;
constexpr int kS  = 2048;
constexpr int kDM = 1024;
constexpr int kH  = 16;
constexpr int kD  = 64;
constexpr int kDI = 1024;    // H*D
constexpr int kBS = kB * kS; // 4096 rows
constexpr float kEps = 1e-5f;
}

// ---------------------------------------------------------------------------
// WMMA fragment helpers (contiguous k order, same for A and B, wave32)
// ---------------------------------------------------------------------------

// 16-bit A-matrix 16x32: lane<16 -> M=lane, K in {0..15}; lane>=16 -> K in {16..31}
__device__ __forceinline__ h16 load_a_frag16x32(const _Float16* A, int lda, int row0, int k0) {
  const int lane = threadIdx.x & 31;
  const int m  = row0 + (lane & 15);
  const int kb = k0 + ((lane < 16) ? 0 : 16);
  const _Float16* p = A + (size_t)m * lda + kb;
  h8v lo = *reinterpret_cast<const h8v*>(p);
  h8v hi = *reinterpret_cast<const h8v*>(p + 8);
  h16 r;
#pragma unroll
  for (int e = 0; e < 8; ++e) { r[e] = lo[e]; r[e + 8] = hi[e]; }
  return r;
}

// B fragment (32x16 K x N) where per-lane 16 K-elements are contiguous in memory.
__device__ __forceinline__ h16 load_b_frag_contig(const _Float16* p) {
  h8v lo = *reinterpret_cast<const h8v*>(p);
  h8v hi = *reinterpret_cast<const h8v*>(p + 8);
  h16 r;
#pragma unroll
  for (int e = 0; e < 8; ++e) { r[e] = lo[e]; r[e + 8] = hi[e]; }
  return r;
}

// B fragment from a K-major ("transposed", [N][K] row-major) matrix
__device__ __forceinline__ h16 load_b_frag_kt(const _Float16* BT, int ldk, int col0, int k0) {
  const int lane = threadIdx.x & 31;
  const _Float16* p = BT + (size_t)(col0 + (lane & 15)) * ldk + k0 + ((lane < 16) ? 0 : 16);
  return load_b_frag_contig(p);
}

__device__ __forceinline__ f8v wmma_f32_f16(h16 a, h16 b, f8v c) {
  return __builtin_amdgcn_wmma_f32_16x16x32_f16(false, a, false, b, (short)0, c, false, false);
}

// ---------------------------------------------------------------------------
// fp32 -> fp16 conversion (plain) and transposing variant (weights -> [N][K])
// ---------------------------------------------------------------------------
__global__ void f32_to_f16_kernel(const float* __restrict__ src, _Float16* __restrict__ dst, int n) {
  for (int i = blockIdx.x * blockDim.x + threadIdx.x; i < n; i += gridDim.x * blockDim.x)
    dst[i] = (_Float16)src[i];
}

__global__ void transpose_f32_to_f16_kernel(const float* __restrict__ src, _Float16* __restrict__ dst,
                                            int R, int C) { // src[R][C] -> dst[C][R]
  for (int i = blockIdx.x * blockDim.x + threadIdx.x; i < R * C; i += gridDim.x * blockDim.x) {
    const int r = i / C, c = i - r * C;
    dst[(size_t)c * R + r] = (_Float16)src[i];
  }
}

// ---------------------------------------------------------------------------
// WMMA GEMM: Y = act(A[MxK] @ B[KxN]) with B given K-major (BT: [N][K]).
// Block = 8 waves -> 128x128 C macro-tile; wave = 64x32 (4 M-tiles x 2 N-tiles).
// Software-pipelined k-loop: next step's 12 x b128 loads are issued before the
// current 8 WMMAs so the compiler can use partial loadcnt waits.
// ACT: 0 none, 1 silu, 2 sigmoid. VTRANS: f16 out as [b,h,d,s] (for V).
// ---------------------------------------------------------------------------
template <int ACT, int VTRANS, bool OUTH, bool OUTF>
__global__ void __launch_bounds__(256, 1)
wmma_gemm_kt_kernel(const _Float16* __restrict__ A,
                    const _Float16* __restrict__ BT,
                    _Float16* __restrict__ outH,
                    float* __restrict__ outF,
                    int M, int N, int K) {
  const int lane = threadIdx.x & 31;
  const int wave = threadIdx.x >> 5;
  const int nbn  = N >> 7;                       // 128-wide N blocks
  const int bm   = blockIdx.x / nbn;
  const int bn   = blockIdx.x - bm * nbn;
  const int row0 = bm * 128 + (wave >> 2) * 64;  // 4 M-tiles
  const int col0 = bn * 128 + (wave & 3) * 32;   // 2 N-tiles
  if (row0 >= M) return;

  f8v acc[4][2] = {};
  h16 aCur[4], bCur[2];
  bCur[0] = load_b_frag_kt(BT, K, col0, 0);
  bCur[1] = load_b_frag_kt(BT, K, col0 + 16, 0);
#pragma unroll
  for (int mt = 0; mt < 4; ++mt) aCur[mt] = load_a_frag16x32(A, K, row0 + mt * 16, 0);

  for (int k0 = 0; k0 < K; k0 += 32) {
    const int k1 = k0 + 32;
    h16 aNxt[4], bNxt[2];
    if (k1 < K) {
      bNxt[0] = load_b_frag_kt(BT, K, col0, k1);
      bNxt[1] = load_b_frag_kt(BT, K, col0 + 16, k1);
#pragma unroll
      for (int mt = 0; mt < 4; ++mt) aNxt[mt] = load_a_frag16x32(A, K, row0 + mt * 16, k1);
    }
#pragma unroll
    for (int mt = 0; mt < 4; ++mt) {
      acc[mt][0] = wmma_f32_f16(aCur[mt], bCur[0], acc[mt][0]);
      acc[mt][1] = wmma_f32_f16(aCur[mt], bCur[1], acc[mt][1]);
    }
    if (k1 < K) {
#pragma unroll
      for (int mt = 0; mt < 4; ++mt) aCur[mt] = aNxt[mt];
      bCur[0] = bNxt[0];
      bCur[1] = bNxt[1];
    }
  }

  // C/D layout: VGPR r -> M = r + (lane<16 ? 0 : 8), N = lane%16
#pragma unroll
  for (int mt = 0; mt < 4; ++mt) {
#pragma unroll
    for (int nt = 0; nt < 2; ++nt) {
      const int mb = row0 + mt * 16 + ((lane < 16) ? 0 : 8);
      const int n  = col0 + nt * 16 + (lane & 15);
#pragma unroll
      for (int r = 0; r < 8; ++r) {
        float v = acc[mt][nt][r];
        if constexpr (ACT == 1) v = v / (1.f + __expf(-v));   // silu
        if constexpr (ACT == 2) v = 1.f / (1.f + __expf(-v)); // sigmoid
        const int m = mb + r;
        if constexpr (OUTH) {
          if constexpr (VTRANS) {
            // m = b*S + s ; n = h*64 + d  ->  [((b*H+h)*64+d)*S + s]
            const int b = m >> 11, s = m & (kS - 1);
            const int h = n >> 6,  d = n & 63;
            outH[(((size_t)b * kH + h) * kD + d) * kS + s] = (_Float16)v;
          } else {
            outH[(size_t)m * N + n] = (_Float16)v;
          }
        }
        if constexpr (OUTF) outF[(size_t)m * N + n] = v;
      }
    }
  }
}

// ---------------------------------------------------------------------------
// log_alpha[b,h,s] = -softplus(x[b,s,:] . Wa[:,h] + ba[h])
// One block per (b,s) row; the 4KB x-row staged into LDS with the CDNA5
// async-to-LDS path, ordered with s_wait_asynccnt.
// ---------------------------------------------------------------------------
__global__ void log_alpha_kernel(const float* __restrict__ x, const float* __restrict__ Wa,
                                 const float* __restrict__ ba, float* __restrict__ la) {
  __shared__ __align__(16) float xrow[kDM];
  __shared__ float part[16][17];
  const int row = blockIdx.x;     // b*S + s
  const int tid = threadIdx.x;    // 256 threads

  {
    const float* gp = x + (size_t)row * kDM + tid * 4;
    unsigned loff = (unsigned)(size_t)(&xrow[tid * 4]);
    asm volatile("global_load_async_to_lds_b128 %0, %1, off"
                 :: "v"(loff), "v"(gp) : "memory");
  }
  asm volatile("s_wait_asynccnt 0x0" ::: "memory");
  __syncthreads();

  const int h  = tid & 15;
  const int pr = tid >> 4;
  float acc = 0.f;
  for (int m = pr; m < kDM; m += 16) acc += xrow[m] * Wa[(size_t)m * kH + h];
  part[pr][h] = acc;
  __syncthreads();

  if (tid < 16) {                 // tid == h
    float s = ba[tid];
#pragma unroll
    for (int i = 0; i < 16; ++i) s += part[i][tid];
    const float sp = (s > 20.f) ? s : log1pf(__expf(s));
    const int b = row / kS, sidx = row - b * kS;
    la[((size_t)b * kH + tid) * kS + sidx] = -sp;
  }
}

// ---------------------------------------------------------------------------
// Inclusive cumsum over s per (b,h): one wave32 per row, shuffle scan.
// ---------------------------------------------------------------------------
__global__ void cumsum_kernel(const float* __restrict__ la, float* __restrict__ cum) {
  const int bh   = blockIdx.x;
  const int lane = threadIdx.x; // blockDim.x == 32
  const float* src = la + (size_t)bh * kS;
  float* dst = cum + (size_t)bh * kS;
  float carry = 0.f;
  for (int s0 = 0; s0 < kS; s0 += 32) {
    float v = src[s0 + lane];
#pragma unroll
    for (int off = 1; off < 32; off <<= 1) {
      float u = __shfl_up(v, off, 32);
      if (lane >= off) v += u;
    }
    v += carry;
    dst[s0 + lane] = v;
    carry = __shfl(v, 31, 32);
  }
}

// ---------------------------------------------------------------------------
// Attention: per (b,h), each wave owns one 16-row output block.
// Branchless causal decay: masked elements get exponent -1e4 -> exp underflows
// to 0 (no EXEC divergence around the WMMA pipeline).
// ---------------------------------------------------------------------------
__global__ void __launch_bounds__(256, 1)
gla_attn_kernel(const _Float16* __restrict__ qh,
                const _Float16* __restrict__ kh,
                const _Float16* __restrict__ vT,
                const float* __restrict__ cum,
                float* __restrict__ o) {
  constexpr int WAVES = 8;
  __shared__ __align__(16) _Float16 ldsP[WAVES][16 * 32];
  const int lane = threadIdx.x & 31;
  const int wave = threadIdx.x >> 5;
  const int bh = blockIdx.y;
  const int b  = bh / kH;
  const int h  = bh - b * kH;
  const int i0 = (blockIdx.x * WAVES + wave) * 16;
  _Float16* myP = &ldsP[wave][0];

  const _Float16* qbase = qh + (size_t)b * kS * kDI + (size_t)h * kD;
  const _Float16* kbase = kh + (size_t)b * kS * kDI + (size_t)h * kD;
  const _Float16* vtb   = vT + (size_t)bh * kD * kS;   // [d][s]
  const float* cumbh = cum + (size_t)bh * kS;

  h16 qa0 = load_a_frag16x32(qbase, kDI, i0, 0);
  h16 qa1 = load_a_frag16x32(qbase, kDI, i0, 32);

  float cumi[8];
#pragma unroll
  for (int r = 0; r < 8; ++r) cumi[r] = cumbh[i0 + r + ((lane < 16) ? 0 : 8)];

  f8v accO[4] = {};

  for (int j0 = 0; j0 <= i0; j0 += 32) {
    // V fragments first: independent of P, overlap with QK^T + decay work
    h16 vb[4];
#pragma unroll
    for (int n = 0; n < 4; ++n) {
      const _Float16* vp = vtb + (size_t)(n * 16 + (lane & 15)) * kS + j0 + ((lane < 16) ? 0 : 16);
      vb[n] = load_b_frag_contig(vp);
    }

#pragma unroll
    for (int t = 0; t < 2; ++t) {
      const int jc = j0 + t * 16;
      const _Float16* kp = kbase + (size_t)(jc + (lane & 15)) * kDI + ((lane < 16) ? 0 : 16);
      f8v sacc = {};
      sacc = wmma_f32_f16(qa0, load_b_frag_contig(kp), sacc);
      sacc = wmma_f32_f16(qa1, load_b_frag_contig(kp + 32), sacc);

      const int   jcol = jc + (lane & 15);
      const float cumj = cumbh[jcol];
#pragma unroll
      for (int r = 0; r < 8; ++r) {
        const int mrow = r + ((lane < 16) ? 0 : 8);
        const int irow = i0 + mrow;
        // branchless causal mask: exp(-1e4) flushes to 0
        const float arg = (jcol <= irow) ? (cumi[r] - cumj) : -1.0e4f;
        const float pv  = sacc[r] * __expf(arg);
        myP[mrow * 32 + t * 16 + (lane & 15)] = (_Float16)pv;
      }
    }
    // order LDS stores before the cross-lane A-fragment reload
    asm volatile("s_wait_dscnt 0x0" ::: "memory");

    h16 pa = load_a_frag16x32(myP, 32, 0, 0);
#pragma unroll
    for (int n = 0; n < 4; ++n) accO[n] = wmma_f32_f16(pa, vb[n], accO[n]);
  }

#pragma unroll
  for (int n = 0; n < 4; ++n) {
#pragma unroll
    for (int r = 0; r < 8; ++r) {
      const int irow = i0 + r + ((lane < 16) ? 0 : 8);
      o[(size_t)(b * kS + irow) * kDI + h * kD + n * 16 + (lane & 15)] = accO[n][r];
    }
  }
}

// ---------------------------------------------------------------------------
// GroupNorm statistics per (b,h) over S x D
// ---------------------------------------------------------------------------
__global__ void gn_stats_kernel(const float* __restrict__ o, float* __restrict__ stats) {
  __shared__ float ssum[256];
  __shared__ float ssq[256];
  const int bh = blockIdx.x;
  const int b = bh / kH, h = bh - b * kH;
  float s = 0.f, q = 0.f;
  const int total = kS * kD;
  for (int idx = threadIdx.x; idx < total; idx += blockDim.x) {
    const int srow = idx >> 6, d = idx & 63;
    const float v = o[(size_t)(b * kS + srow) * kDI + h * kD + d];
    s += v; q += v * v;
  }
  ssum[threadIdx.x] = s; ssq[threadIdx.x] = q;
  __syncthreads();
  for (int st = 128; st > 0; st >>= 1) {
    if ((int)threadIdx.x < st) {
      ssum[threadIdx.x] += ssum[threadIdx.x + st];
      ssq[threadIdx.x]  += ssq[threadIdx.x + st];
    }
    __syncthreads();
  }
  if (threadIdx.x == 0) {
    const float inv  = 1.f / (float)total;
    const float mean = ssum[0] * inv;
    const float var  = ssq[0] * inv - mean * mean;
    stats[bh * 2]     = mean;
    stats[bh * 2 + 1] = rsqrtf(var + kEps);
  }
}

// ---------------------------------------------------------------------------
// og = ((o - mean) * rstd * gn_w + gn_b) * gate   (f16 out for final GEMM)
// ---------------------------------------------------------------------------
__global__ void gn_gate_kernel(const float* __restrict__ o, const _Float16* __restrict__ gate,
                               const float* __restrict__ stats,
                               const float* __restrict__ gw, const float* __restrict__ gb,
                               _Float16* __restrict__ og) {
  const size_t total = (size_t)kBS * kDI;
  for (size_t i = (size_t)blockIdx.x * blockDim.x + threadIdx.x; i < total;
       i += (size_t)gridDim.x * blockDim.x) {
    const int c = (int)(i % kDI);
    const size_t row = i / kDI;   // b*S + s
    const int b = (int)(row / kS);
    const int h = c >> 6;
    const int bh = b * kH + h;
    const float v = (o[i] - stats[bh * 2]) * stats[bh * 2 + 1] * gw[c] + gb[c];
    og[i] = (_Float16)(v * (float)gate[i]);
  }
}

// ---------------------------------------------------------------------------
// Host launcher
// ---------------------------------------------------------------------------
extern "C" void kernel_launch(void* const* d_in, const int* /*in_sizes*/, int /*n_in*/,
                              void* d_out, int /*out_size*/, void* d_ws, size_t /*ws_size*/,
                              hipStream_t stream) {
  const float* x  = (const float*)d_in[0];
  const float* Wq = (const float*)d_in[1];
  const float* Wk = (const float*)d_in[2];
  const float* Wv = (const float*)d_in[3];
  const float* Wg = (const float*)d_in[4];
  const float* Wa = (const float*)d_in[5];
  const float* ba = (const float*)d_in[6];
  const float* Wo = (const float*)d_in[7];
  const float* gw = (const float*)d_in[8];
  const float* gb = (const float*)d_in[9];
  float* out = (float*)d_out;

  // bump-allocate workspace (256B aligned slices)
  char* p = (char*)d_ws;
  auto bump = [&](size_t bytes) -> char* {
    char* r = p;
    p += (bytes + 255) & ~(size_t)255;
    return r;
  };
  _Float16* xh    = (_Float16*)bump((size_t)kBS * kDM * 2);
  _Float16* WqT   = (_Float16*)bump((size_t)kDM * kDI * 2);  // [N][K]
  _Float16* WkT   = (_Float16*)bump((size_t)kDM * kDI * 2);
  _Float16* WvT   = (_Float16*)bump((size_t)kDM * kDI * 2);
  _Float16* WgT   = (_Float16*)bump((size_t)kDM * kDI * 2);
  _Float16* WoT   = (_Float16*)bump((size_t)kDI * kDM * 2);
  _Float16* qhd   = (_Float16*)bump((size_t)kBS * kDI * 2);
  _Float16* khd   = (_Float16*)bump((size_t)kBS * kDI * 2);
  _Float16* vTd   = (_Float16*)bump((size_t)kB * kH * kD * kS * 2); // [b,h,d,s]
  _Float16* gateh = (_Float16*)bump((size_t)kBS * kDI * 2);
  float*    la    = (float*)bump((size_t)kB * kH * kS * 4);
  float*    cumv  = (float*)bump((size_t)kB * kH * kS * 4);
  float*    ov    = (float*)bump((size_t)kBS * kDI * 4);
  float*    stats = (float*)bump((size_t)kB * kH * 2 * 4);
  _Float16* og    = (_Float16*)bump((size_t)kBS * kDI * 2);

  const int T = 256;

  // 1) conversions: x plain; weights transposed to K-major for WMMA B feed
  f32_to_f16_kernel<<<512, T, 0, stream>>>(x, xh, kBS * kDM);
  transpose_f32_to_f16_kernel<<<256, T, 0, stream>>>(Wq, WqT, kDM, kDI);
  transpose_f32_to_f16_kernel<<<256, T, 0, stream>>>(Wk, WkT, kDM, kDI);
  transpose_f32_to_f16_kernel<<<256, T, 0, stream>>>(Wv, WvT, kDM, kDI);
  transpose_f32_to_f16_kernel<<<256, T, 0, stream>>>(Wg, WgT, kDM, kDI);
  transpose_f32_to_f16_kernel<<<256, T, 0, stream>>>(Wo, WoT, kDI, kDM);

  // 2) projections (software-pipelined WMMA GEMM + fused activation)
  const int gemmBlocks = (kBS / 128) * (kDI / 128); // 32 * 8 = 256
  wmma_gemm_kt_kernel<1, 0, true, false><<<gemmBlocks, T, 0, stream>>>(xh, WqT, qhd,   nullptr, kBS, kDI, kDM);
  wmma_gemm_kt_kernel<1, 0, true, false><<<gemmBlocks, T, 0, stream>>>(xh, WkT, khd,   nullptr, kBS, kDI, kDM);
  wmma_gemm_kt_kernel<0, 1, true, false><<<gemmBlocks, T, 0, stream>>>(xh, WvT, vTd,   nullptr, kBS, kDI, kDM);
  wmma_gemm_kt_kernel<2, 0, true, false><<<gemmBlocks, T, 0, stream>>>(xh, WgT, gateh, nullptr, kBS, kDI, kDM);

  // 3) decay gates + cumulative sum
  log_alpha_kernel<<<kBS, T, 0, stream>>>(x, Wa, ba, la);
  cumsum_kernel<<<kB * kH, 32, 0, stream>>>(la, cumv);

  // 4) decayed quadratic attention (WMMA both matmuls)
  gla_attn_kernel<<<dim3(kS / 16 / 8, kB * kH), T, 0, stream>>>(qhd, khd, vTd, cumv, ov);

  // 5) group norm + gate
  gn_stats_kernel<<<kB * kH, T, 0, stream>>>(ov, stats);
  gn_gate_kernel<<<2048, T, 0, stream>>>(ov, gateh, stats, gw, gb, og);

  // 6) output projection (WMMA GEMM, f32 output)
  wmma_gemm_kt_kernel<0, 0, false, true><<<gemmBlocks, T, 0, stream>>>(og, WoT, nullptr, out, kBS, kDM, kDI);
}
